// Dynamics_15599321219162
// MI455X (gfx1250) — matrix operand order repaired and checked by reading, not compiled, not run on hardware
//
#include <hip/hip_runtime.h>

// ---- CDNA5 WMMA types -------------------------------------------------------
typedef __attribute__((ext_vector_type(16))) __bf16       v16bf;
typedef __attribute__((ext_vector_type(8)))  float        v8f;
typedef __attribute__((ext_vector_type(4)))  unsigned int u32x4;

union Frag {
    v16bf v;
    u32x4 q[2];
};

#define BM 128
#define BN 128
#define BK 32
#define THREADS 256
#define LDS_STRIDE (BK + 8)   // 40 bf16 = 80B row stride, keeps 16B alignment

// ---- prep kernels -----------------------------------------------------------
__global__ void zero_counts_kernel(int* wsI) {
    if (threadIdx.x < 16) wsI[threadIdx.x] = 0;   // counts[8] + cursors[8]
}

__global__ void hist_kernel(const int* __restrict__ idx, int* __restrict__ counts, int B) {
    int b = blockIdx.x * blockDim.x + threadIdx.x;
    if (b < B) atomicAdd(&counts[idx[b] & 7], 1);
}

__global__ void scan_kernel(int* wsI) {
    if (threadIdx.x == 0) {
        int off = 0;
        for (int p = 0; p < 8; ++p) {
            wsI[16 + p] = off;   // offsets[p]
            wsI[8 + p]  = off;   // cursors[p]
            off += wsI[p];
        }
        wsI[24] = off;           // offsets[8] == B
    }
}

__global__ void scatter_kernel(const int* __restrict__ idx, int* __restrict__ cursors,
                               int* __restrict__ order, int B) {
    int b = blockIdx.x * blockDim.x + threadIdx.x;
    if (b < B) {
        int pos = atomicAdd(&cursors[idx[b] & 7], 1);
        order[pos] = b;
    }
}

// Pack concat(latent, action) rows in bucketed order -> bf16 [B][544], zero pad 528..543
__global__ void pack_kernel(const float* __restrict__ lat, const float* __restrict__ act,
                            const int* __restrict__ order, __bf16* __restrict__ Xp) {
    int r = blockIdx.x;
    int s = order[r];
    for (int c = threadIdx.x; c < 544; c += blockDim.x) {
        float v = 0.0f;
        if (c < 512)      v = lat[(unsigned)s * 512u + (unsigned)c];
        else if (c < 528) v = act[(unsigned)s * 16u + (unsigned)(c - 512)];
        Xp[(unsigned)r * 544u + (unsigned)c] = (__bf16)v;
    }
}

// Zero-pad W1: [8][528][1024] fp32 -> [8][544][1024] fp32 (rows 528..543 = 0)
__global__ void pad_w1_kernel(const float* __restrict__ W1, float* __restrict__ W1p) {
    unsigned e = blockIdx.x * blockDim.x + threadIdx.x;   // 8*544*1024 threads
    const unsigned KN = 544u * 1024u;
    unsigned p   = e / KN;
    unsigned rem = e - p * KN;
    unsigned k   = rem >> 10;
    unsigned n   = rem & 1023u;
    float v = (k < 528u) ? W1[(p * 528u + k) * 1024u + n] : 0.0f;
    W1p[e] = v;
}

// ---- grouped expert GEMM: C[m,n] = act( A[m,:] * W[:,n] + bias[n] ) ---------
// A : bf16 packed rows (bucketed order), LDA elems/row (multiple of 32)
// W : fp32 [KTILES*32][N] per expert (pre-padded) -> bf16 on the fly
// FINAL==false: relu + bf16 store to outB (bucketed rows)
// FINAL==true : fp32 scatter to outF[order[row]*N + col], no relu
template<int KTILES, int N, int LDA, bool FINAL>
__global__ __launch_bounds__(THREADS)
void gemm_expert_kernel(const __bf16* __restrict__ A,
                        const float* __restrict__ Wg,
                        const float* __restrict__ biasg,
                        __bf16* __restrict__ outB,
                        float* __restrict__ outF,
                        const int* __restrict__ offsets,
                        const int* __restrict__ order,
                        int Btot)
{
    __shared__ __bf16 sA [BM][LDS_STRIDE];   // A tile  [row][k], 10KB
    __shared__ __bf16 sBt[BN][LDS_STRIDE];   // W tileT [col][k], 10KB

    const int p    = blockIdx.z;
    const int base = offsets[p];
    const int cnt  = offsets[p + 1] - base;
    const int m0   = blockIdx.y * BM;
    if (m0 >= cnt) return;                    // block-uniform: EXEC stays all-ones
    const int n0   = blockIdx.x * BN;

    const float* W    = Wg    + (size_t)p * (size_t)(KTILES * BK * N);
    const float* bias = biasg + p * N;

    const int tid  = threadIdx.x;
    const int lane = tid & 31;
    const int wave = tid >> 5;
    const int wm   = wave & 1;    // 0..1 -> 64-row halves
    const int wn   = wave >> 1;   // 0..3 -> 32-col quarters

    const v8f vzero = {0.f, 0.f, 0.f, 0.f, 0.f, 0.f, 0.f, 0.f};
    v8f acc[4][2];
#pragma unroll
    for (int i = 0; i < 4; ++i)
#pragma unroll
        for (int j = 0; j < 2; ++j) acc[i][j] = vzero;

    // A staging: thread -> (row, 16-elem segment); advancing pointer, imm offsets
    const int arow = tid >> 1;
    const int aseg = (tid & 1) * 16;
    int argl = base + m0 + arow;
    if (argl >= Btot) argl = Btot - 1;        // clamp (stores are guarded by cnt)
    const __bf16* aP = A + (size_t)argl * LDA + aseg;

    // W staging: thread -> (col, 16-row K chunk); coalesced fp32, imm offsets kk*N
    const int wcol = tid & 127;
    const int wkq  = (tid >> 7) * 16;
    const float* wP = W + wkq * N + (n0 + wcol);

    // pipeline registers (tile kt+1 loads overlap tile kt compute)
    u32x4 aR0, aR1;
    float wR[16];
    {
        const u32x4* ag = (const u32x4*)aP;
        aR0 = ag[0];
        aR1 = ag[1];
#pragma unroll
        for (int kk = 0; kk < 16; ++kk) wR[kk] = wP[kk * N];
    }

    const int arl   = lane & 15;
    const int akoff = (lane >> 4) * 8;
    const int bkoff = (lane >> 4) * 8;

    for (int kt = 0; kt < KTILES; ++kt) {
        // commit staged registers to LDS
        *(u32x4*)&sA[arow][aseg]     = aR0;
        *(u32x4*)&sA[arow][aseg + 8] = aR1;
        {
            __bf16 tmp[16];
#pragma unroll
            for (int kk = 0; kk < 16; ++kk) tmp[kk] = (__bf16)wR[kk];
            *(u32x4*)&sBt[wcol][wkq]     = *(u32x4*)&tmp[0];
            *(u32x4*)&sBt[wcol][wkq + 8] = *(u32x4*)&tmp[8];
        }
        __syncthreads();

        // prefetch next tile into registers (overlaps ds_loads + wmma below)
        if (kt + 1 < KTILES) {
            aP += BK;
            wP += BK * N;
            const u32x4* ag = (const u32x4*)aP;
            aR0 = ag[0];
            aR1 = ag[1];
#pragma unroll
            for (int kk = 0; kk < 16; ++kk) wR[kk] = wP[kk * N];
        }

        // A fragments: lanes 0-15 row m, K {0..7,16..23}; lanes 16-31 K {8..15,24..31}
        Frag afr[4];
#pragma unroll
        for (int ms = 0; ms < 4; ++ms) {
            const int r = wm * 64 + ms * 16 + arl;
            afr[ms].q[0] = *(const u32x4*)&sA[r][akoff];
            afr[ms].q[1] = *(const u32x4*)&sA[r][16 + akoff];
        }
        // B fragments from W^T rows: lanes 0-15 col n, K {0..7,16..23}; lanes 16-31 K {8..15,24..31}
        Frag bfr[2];
#pragma unroll
        for (int ns = 0; ns < 2; ++ns) {
            const int c = wn * 32 + ns * 16 + arl;
            bfr[ns].q[0] = *(const u32x4*)&sBt[c][bkoff];
            bfr[ns].q[1] = *(const u32x4*)&sBt[c][16 + bkoff];
        }

#pragma unroll
        for (int ms = 0; ms < 4; ++ms)
#pragma unroll
            for (int ns = 0; ns < 2; ++ns)
                acc[ms][ns] = __builtin_amdgcn_wmma_f32_16x16x32_bf16(
                    false, afr[ms].v, false, bfr[ns].v,
                    (short)0, acc[ms][ns], false, false);

        __syncthreads();
    }

    // epilogue: C VGPR r -> row (r + 8*(lane>=16)), col (lane&15)
    const int cl    = lane & 15;
    const int rhalf = (lane >> 4) * 8;
    const int rbase = m0 + wm * 64 + rhalf;     // + ms*16 + r
    const bool full = (m0 + BM <= cnt);         // uniform: interior tile, no guards

    if (full) {
#pragma unroll
        for (int ns = 0; ns < 2; ++ns) {
            const int col = n0 + wn * 32 + ns * 16 + cl;
            const float bv = bias[col];
#pragma unroll
            for (int ms = 0; ms < 4; ++ms) {
#pragma unroll
                for (int r = 0; r < 8; ++r) {
                    const int rowl = rbase + ms * 16 + r;
                    float v = acc[ms][ns][r] + bv;
                    if (FINAL) {
                        const int s = order[base + rowl];
                        outF[(unsigned)s * (unsigned)N + (unsigned)col] = v;
                    } else {
                        v = v > 0.f ? v : 0.f;
                        outB[(unsigned)(base + rowl) * (unsigned)N + (unsigned)col] = (__bf16)v;
                    }
                }
            }
        }
    } else {
#pragma unroll
        for (int ns = 0; ns < 2; ++ns) {
            const int col = n0 + wn * 32 + ns * 16 + cl;
            const float bv = bias[col];
#pragma unroll
            for (int ms = 0; ms < 4; ++ms) {
#pragma unroll
                for (int r = 0; r < 8; ++r) {
                    const int rowl = rbase + ms * 16 + r;
                    if (rowl < cnt) {
                        float v = acc[ms][ns][r] + bv;
                        if (FINAL) {
                            const int s = order[base + rowl];
                            outF[(unsigned)s * (unsigned)N + (unsigned)col] = v;
                        } else {
                            v = v > 0.f ? v : 0.f;
                            outB[(unsigned)(base + rowl) * (unsigned)N + (unsigned)col] = (__bf16)v;
                        }
                    }
                }
            }
        }
    }
}

// ---- launch -----------------------------------------------------------------
extern "C" void kernel_launch(void* const* d_in, const int* in_sizes, int n_in,
                              void* d_out, int out_size, void* d_ws, size_t ws_size,
                              hipStream_t stream)
{
    const float* latents = (const float*)d_in[0];
    const float* actions = (const float*)d_in[1];
    const int*   pidx    = (const int*)  d_in[2];
    const float* W1      = (const float*)d_in[3];
    const float* b1      = (const float*)d_in[4];
    const float* W2      = (const float*)d_in[5];
    const float* b2      = (const float*)d_in[6];
    const float* W3      = (const float*)d_in[7];
    const float* b3      = (const float*)d_in[8];
    float* out = (float*)d_out;

    const int B = 4096, DLAT = 512, H = 1024;
    const int DPAD = 544;              // K padded to multiple of 32 (528 -> 544)

    // workspace layout
    int* wsI     = (int*)d_ws;         // [0..7] counts, [8..15] cursors, [16..24] offsets
    int* counts  = wsI;
    int* cursors = wsI + 8;
    int* offsets = wsI + 16;
    int* order   = wsI + 32;           // [B]
    char* wsB = (char*)d_ws;
    size_t off = 16640;                // 256B-aligned past the int region
    __bf16* Xp  = (__bf16*)(wsB + off); off += (size_t)B * DPAD * sizeof(__bf16);
    __bf16* H1  = (__bf16*)(wsB + off); off += (size_t)B * H * sizeof(__bf16);
    __bf16* H2  = (__bf16*)(wsB + off); off += (size_t)B * H * sizeof(__bf16);
    float*  W1p = (float*)(wsB + off);  // [8][544][1024] fp32, zero-padded K rows
    (void)ws_size; (void)in_sizes; (void)n_in; (void)out_size;

    zero_counts_kernel<<<1, 32, 0, stream>>>(wsI);
    hist_kernel<<<B / 256, 256, 0, stream>>>(pidx, counts, B);
    scan_kernel<<<1, 32, 0, stream>>>(wsI);
    scatter_kernel<<<B / 256, 256, 0, stream>>>(pidx, cursors, order, B);
    pack_kernel<<<B, 256, 0, stream>>>(latents, actions, order, Xp);
    pad_w1_kernel<<<(8 * DPAD * 1024) / 256, 256, 0, stream>>>(W1, W1p);

    dim3 blk(THREADS);
    // layer 1: [cnt x 544] * [544 x 1024] -> relu -> H1
    gemm_expert_kernel<DPAD / BK, 1024, DPAD, false>
        <<<dim3(H / BN, B / BM, 8), blk, 0, stream>>>(
            Xp, W1p, b1, H1, nullptr, offsets, order, B);
    // layer 2: [cnt x 1024] * [1024 x 1024] -> relu -> H2
    gemm_expert_kernel<H / BK, 1024, 1024, false>
        <<<dim3(H / BN, B / BM, 8), blk, 0, stream>>>(
            H1, W2, b2, H2, nullptr, offsets, order, B);
    // layer 3: [cnt x 1024] * [1024 x 512] + b3 -> scatter fp32 to d_out
    gemm_expert_kernel<H / BK, 512, 1024, true>
        <<<dim3(DLAT / BN, B / BM, 8), blk, 0, stream>>>(
            H2, W3, b3, nullptr, out, offsets, order, B);
}
